// PWCLONet_3169685865250
// MI455X (gfx1250) — matrix operand order repaired and checked by reading, not compiled, not run on hardware
//
#include <hip/hip_runtime.h>
#include <hip/hip_bf16.h>
#include <math.h>

// ============================ CDNA5 WMMA types ============================
typedef __attribute__((ext_vector_type(16))) _Float16 v16h;
typedef __attribute__((ext_vector_type(8)))  _Float16 v8h;
typedef __attribute__((ext_vector_type(8)))  float    v8f;

#define FPS_T 512

static inline int g1i(int n, int t) { return (n + t - 1) / t; }
static inline int pad32(int x) { return (x + 31) & ~31; }

// ============================ elementwise / point kernels ============================

// (B,3,N) -> (B,N,3)
__global__ void k_transpose(const float* __restrict__ in, int B, int N, float* __restrict__ out) {
  int i = blockIdx.x * blockDim.x + threadIdx.x;
  if (i >= B * N) return;
  int b = i / N, n = i % N;
  const float* ip = in + (size_t)b * 3 * N;
  float* op = out + (size_t)i * 3;
  op[0] = ip[n]; op[1] = ip[(size_t)N + n]; op[2] = ip[(size_t)2 * N + n];
}

// Farthest point sampling: one block per batch; LDS-resident distance array.
__global__ void k_fps(const float* __restrict__ xyz, int N, int npoint, int* __restrict__ out) {
  __shared__ float dist[8192];
  __shared__ float rv[FPS_T];
  __shared__ int   ri[FPS_T];
  __shared__ float lp[3];
  int b = blockIdx.x;
  const float* px = xyz + (size_t)b * N * 3;
  int t = threadIdx.x;
  for (int i = t; i < N; i += FPS_T) dist[i] = 1e10f;
  int last = 0;
  if (t == 0) out[(size_t)b * npoint] = 0;
  __syncthreads();
  for (int it = 1; it < npoint; ++it) {
    if (t == 0) { lp[0] = px[(size_t)last*3]; lp[1] = px[(size_t)last*3+1]; lp[2] = px[(size_t)last*3+2]; }
    __syncthreads();
    float bv = -1.f; int bi = 0;
    for (int i = t; i < N; i += FPS_T) {
      float dx = px[(size_t)i*3]   - lp[0];
      float dy = px[(size_t)i*3+1] - lp[1];
      float dz = px[(size_t)i*3+2] - lp[2];
      float d = dx*dx + dy*dy + dz*dz;
      float nd = fminf(dist[i], d);
      dist[i] = nd;
      if (nd > bv) { bv = nd; bi = i; }
    }
    rv[t] = bv; ri[t] = bi;
    __syncthreads();
    for (int s2 = FPS_T / 2; s2 > 0; s2 >>= 1) {
      if (t < s2) {
        if (rv[t+s2] > rv[t] || (rv[t+s2] == rv[t] && ri[t+s2] < ri[t])) { rv[t] = rv[t+s2]; ri[t] = ri[t+s2]; }
      }
      __syncthreads();
    }
    last = ri[0];
    if (t == 0) out[(size_t)b * npoint + it] = last;
    __syncthreads();
  }
}

// Brute-force k-NN (k <= 32), one thread per query, sorted insertion.
__global__ void k_knn(const float* __restrict__ q, const float* __restrict__ r,
                      int B, int M, int N, int k, int* __restrict__ out) {
  int i = blockIdx.x * blockDim.x + threadIdx.x;
  if (i >= B * M) return;
  int b = i / M;
  const float* qp = q + (size_t)i * 3;
  float qx = qp[0], qy = qp[1], qz = qp[2];
  const float* rp = r + (size_t)b * N * 3;
  float bd[32]; int bi[32];
  for (int x = 0; x < k; ++x) { bd[x] = 3.0e38f; bi[x] = 0; }
  for (int j = 0; j < N; ++j) {
    if (j + 64 < N) __builtin_prefetch(rp + (size_t)(j + 64) * 3, 0, 0);
    float dx = rp[(size_t)j*3]   - qx;
    float dy = rp[(size_t)j*3+1] - qy;
    float dz = rp[(size_t)j*3+2] - qz;
    float d = dx*dx + dy*dy + dz*dz;
    if (d < bd[k-1]) {
      int p = k - 1;
      while (p > 0 && bd[p-1] > d) { bd[p] = bd[p-1]; bi[p] = bi[p-1]; --p; }
      bd[p] = d; bi[p] = j;
    }
  }
  int* op = out + (size_t)i * k;
  for (int x = 0; x < k; ++x) op[x] = bi[x];
}

// out[b,m,:] = in[b, idx[b,m], :]
__global__ void k_gather_rows(const float* __restrict__ in, const int* __restrict__ idx,
                              int B, int Nin, int M, int C, float* __restrict__ out) {
  int i = blockIdx.x * blockDim.x + threadIdx.x;
  if (i >= B * M) return;
  int b = i / M;
  int j = idx[i];
  const float* ip = in + ((size_t)b * Nin + j) * C;
  float* op = out + (size_t)i * C;
  for (int c = 0; c < C; ++c) op[c] = ip[c];
}

// out[b,m,s,:] = [ xyz[idx]-newx[b,m]  (3) , feats[idx] (C) ]
__global__ void k_group_concat(const float* __restrict__ xyz, const float* __restrict__ newx,
                               const int* __restrict__ idx, const float* __restrict__ feats, int C,
                               int B, int M, int S, int Nref, float* __restrict__ out) {
  int i = blockIdx.x * blockDim.x + threadIdx.x;
  if (i >= B * M * S) return;
  int bm = i / S;
  int b = bm / M;
  int j = idx[i];
  const float* xp = xyz + ((size_t)b * Nref + j) * 3;
  const float* np_ = newx + (size_t)bm * 3;
  float* op = out + (size_t)i * (3 + C);
  op[0] = xp[0] - np_[0]; op[1] = xp[1] - np_[1]; op[2] = xp[2] - np_[2];
  if (C > 0) {
    const float* fp = feats + ((size_t)b * Nref + j) * C;
    for (int c = 0; c < C; ++c) op[3 + c] = fp[c];
  }
}

// cost-volume concat: [ f1[b,m] (C) , f2[b,idx] (C) , xyz2[idx]-xyz1[b,m] (3) ]
__global__ void k_cv_concat(const float* __restrict__ f1, const float* __restrict__ f2,
                            const float* __restrict__ xyz1, const float* __restrict__ xyz2,
                            const int* __restrict__ idx, int B, int M, int S, int Nref, int C,
                            float* __restrict__ out) {
  int i = blockIdx.x * blockDim.x + threadIdx.x;
  if (i >= B * M * S) return;
  int bm = i / S;
  int b = bm / M;
  int j = idx[i];
  float* op = out + (size_t)i * (2 * C + 3);
  const float* f1p = f1 + (size_t)bm * C;
  const float* f2p = f2 + ((size_t)b * Nref + j) * C;
  for (int c = 0; c < C; ++c) op[c] = f1p[c];
  for (int c = 0; c < C; ++c) op[C + c] = f2p[c];
  const float* xq = xyz1 + (size_t)bm * 3;
  const float* xr = xyz2 + ((size_t)b * Nref + j) * 3;
  op[2*C]   = xr[0] - xq[0];
  op[2*C+1] = xr[1] - xq[1];
  op[2*C+2] = xr[2] - xq[2];
}

// generic row-wise concat of up to 3 tensors
__global__ void k_concat_rows(const float* __restrict__ a, int Ca,
                              const float* __restrict__ b, int Cb,
                              const float* __restrict__ c, int Cc,
                              int R, float* __restrict__ out) {
  int i = blockIdx.x * blockDim.x + threadIdx.x;
  if (i >= R) return;
  float* op = out + (size_t)i * (Ca + Cb + Cc);
  for (int x = 0; x < Ca; ++x) op[x] = a[(size_t)i * Ca + x];
  for (int x = 0; x < Cb; ++x) op[Ca + x] = b[(size_t)i * Cb + x];
  for (int x = 0; x < Cc; ++x) op[Ca + Cb + x] = c[(size_t)i * Cc + x];
}

// max over S: (B,M,S,C) -> (B,M,C)
__global__ void k_maxpool(const float* __restrict__ in, int B, int M, int S, int C,
                          float* __restrict__ out) {
  int i = blockIdx.x * blockDim.x + threadIdx.x;
  if (i >= B * M * C) return;
  int c = i % C; int bm = i / C;
  const float* ip = in + ((size_t)bm * S) * C + c;
  float v = -3.4e38f;
  for (int s = 0; s < S; ++s) v = fmaxf(v, ip[(size_t)s * C]);
  out[i] = v;
}

// out[b,m,c] = sum_s softmax_s(wlog[b,m,s,c]) * feat[b,m,s,c]
__global__ void k_softmax_wsum(const float* __restrict__ wlog, const float* __restrict__ feat,
                               int B, int M, int S, int C, float* __restrict__ out) {
  int i = blockIdx.x * blockDim.x + threadIdx.x;
  if (i >= B * M * C) return;
  int c = i % C; int bm = i / C;
  const float* wp = wlog + ((size_t)bm * S) * C + c;
  const float* fp = feat + ((size_t)bm * S) * C + c;
  float mx = -3.4e38f;
  for (int s = 0; s < S; ++s) mx = fmaxf(mx, wp[(size_t)s * C]);
  float se = 0.f, acc = 0.f;
  for (int s = 0; s < S; ++s) {
    float e = expf(wp[(size_t)s * C] - mx);
    se += e; acc += e * fp[(size_t)s * C];
  }
  out[i] = acc / se;
}

// pose head pooling: pooled[b,c] = sum_n feats[b,n,c]*softmax_n(mask[b,n,c])
__global__ void k_masked_pool(const float* __restrict__ feats, const float* __restrict__ mask,
                              int B, int M, int C, float* __restrict__ out) {
  int i = blockIdx.x * blockDim.x + threadIdx.x;
  if (i >= B * C) return;
  int c = i % C; int b = i / C;
  const float* mp = mask  + ((size_t)b * M) * C + c;
  const float* fp = feats + ((size_t)b * M) * C + c;
  float mx = -3.4e38f;
  for (int n = 0; n < M; ++n) mx = fmaxf(mx, mp[(size_t)n * C]);
  float se = 0.f, acc = 0.f;
  for (int n = 0; n < M; ++n) {
    float e = expf(mp[(size_t)n * C] - mx);
    se += e; acc += e * fp[(size_t)n * C];
  }
  out[i] = acc / se;
}

// inverse-distance interpolation with 3 neighbors
__global__ void k_upsample(const float* __restrict__ xf, const float* __restrict__ xc,
                           const int* __restrict__ idx, const float* __restrict__ fc_,
                           int B, int Nf, int Nc, int C, float* __restrict__ out) {
  int i = blockIdx.x * blockDim.x + threadIdx.x;
  if (i >= B * Nf) return;
  int b = i / Nf;
  const float* qp = xf + (size_t)i * 3;
  const int* ip = idx + (size_t)i * 3;
  float w[3]; float tw = 0.f;
  for (int j = 0; j < 3; ++j) {
    const float* cp = xc + ((size_t)b * Nc + ip[j]) * 3;
    float dx = cp[0]-qp[0], dy = cp[1]-qp[1], dz = cp[2]-qp[2];
    float d = dx*dx + dy*dy + dz*dz;
    w[j] = 1.f / (d + 1e-8f);
    tw += w[j];
  }
  float* op = out + (size_t)i * C;
  for (int c = 0; c < C; ++c) {
    float a = 0.f;
    for (int j = 0; j < 3; ++j) a += (w[j] / tw) * fc_[((size_t)b * Nc + ip[j]) * C + c];
    op[c] = a;
  }
}

// tiny dense layer (pose heads): Y[r,c] = act(X[r,:]@W + b)
__global__ void k_linear_small(const float* __restrict__ X, const float* __restrict__ W,
                               const float* __restrict__ bias, float* __restrict__ Y,
                               int R, int K, int C, int relu) {
  int i = blockIdx.x * blockDim.x + threadIdx.x;
  if (i >= R * C) return;
  int r = i / C, c = i % C;
  float a = bias ? bias[c] : 0.f;
  for (int k = 0; k < K; ++k) a += X[(size_t)r * K + k] * W[(size_t)k * C + c];
  if (relu) a = fmaxf(a, 0.f);
  Y[i] = a;
}

// ============================ packed-f16 WMMA GEMM path ============================
// Pack activations: X (R x K f32, leading dim ldx) -> P (R x Kpad f16, zero-padded K).
__global__ void k_pack_x(const float* __restrict__ X, int ldx, int R, int K, int Kpad,
                         _Float16* __restrict__ P) {
  int i = blockIdx.x * blockDim.x + threadIdx.x;
  if (i >= R * Kpad) return;
  int r = i / Kpad, k = i % Kpad;
  P[i] = (k < K) ? (_Float16)X[(size_t)r * ldx + k] : (_Float16)0.f;
}

// Pack weights column-major: W (K x Cout f32) -> Wp (Cpad x Kpad f16, zero-padded),
// bias -> bp (Cpad f32, zero-padded).
__global__ void k_pack_w(const float* __restrict__ W, const float* __restrict__ b,
                         int K, int Cout, int Kpad, int Cpad,
                         _Float16* __restrict__ Wp, float* __restrict__ bp) {
  int i = blockIdx.x * blockDim.x + threadIdx.x;
  if (i < Cpad) bp[i] = (i < Cout) ? b[i] : 0.f;
  if (i >= Cpad * Kpad) return;
  int n = i / Kpad, k = i % Kpad;
  Wp[i] = (k < K && n < Cout) ? (_Float16)W[(size_t)k * Cout + n] : (_Float16)0.f;
}

// Fused linear on packed f16 operands. One wave = one 16x16 C-tile.
// Inner loop: 4x global_load_b128 + v_wmma_f32_16x16x32_f16, fully unrolled (KPAD known).
// Fragment layouts per CDNA5 ISA 7.12.2 (wave32):
//   A (16x32 f16): lane m=lane&15, halves 0..7 -> K=8*(lane>>4)+i, 8..15 -> K=16+8*(lane>>4)+i
//   B (32x16 f16): lane n=lane&15, halves 0..7 -> K=8*(lane>>4)+i, 8..15 -> K=16+8*(lane>>4)+i   (Wp is column-major)
//   C/D f32:       VGPR r -> M = r + 8*(lane>>4), N = lane&15
template<int KPAD>
__device__ __forceinline__ void wmma_lin_body(
    const _Float16* __restrict__ Xp, const _Float16* __restrict__ Wp,
    const float* __restrict__ biasp, int R, int Cout, int relu,
    float* __restrict__ Yf, _Float16* __restrict__ Yh, int ldo, int kpad_rt) {
  const int wave = threadIdx.x >> 5, lane = threadIdx.x & 31;
  const int row0 = (blockIdx.y * 2 + wave) * 16;
  const int col0 = blockIdx.x * 16;
  if (row0 >= R) return;                         // wave-uniform
  const int kpad = (KPAD > 0) ? KPAD : kpad_rt;
  const int mA = lane & 15, kGrpA = (lane >> 4) * 8;
  const int nB = lane & 15, kGrpB = (lane >> 4) * 8;
  const _Float16* xr = Xp + (size_t)(row0 + mA) * kpad;
  const _Float16* wc = Wp + (size_t)(col0 + nB) * kpad;
  v8f acc = {};
#pragma unroll
  for (int k0 = 0; k0 < ((KPAD > 0) ? KPAD : 4096); k0 += 32) {
    if (KPAD <= 0 && k0 >= kpad) break;
    v8h alo = *(const v8h*)(xr + k0 + kGrpA);
    v8h ahi = *(const v8h*)(xr + k0 + 16 + kGrpA);
    v8h blo = *(const v8h*)(wc + k0 + kGrpB);
    v8h bhi = *(const v8h*)(wc + k0 + 16 + kGrpB);
    v16h a = __builtin_shufflevector(alo, ahi, 0,1,2,3,4,5,6,7,8,9,10,11,12,13,14,15);
    v16h b = __builtin_shufflevector(blo, bhi, 0,1,2,3,4,5,6,7,8,9,10,11,12,13,14,15);
    acc = __builtin_amdgcn_wmma_f32_16x16x32_f16(false, a, false, b,
                                                 (short)0, acc, false, false);
  }
  const int col = col0 + nB;
  const float bv = biasp[col];
#pragma unroll
  for (int r = 0; r < 8; ++r) {
    int row = row0 + (lane >> 4) * 8 + r;
    float y = acc[r] + bv;
    if (relu) y = fmaxf(y, 0.f);
    if (Yf) {
      if (row < R && col < Cout) Yf[(size_t)row * ldo + col] = y;
    } else {
      if (row < R) Yh[(size_t)row * ldo + col] = (col < Cout) ? (_Float16)y : (_Float16)0.f;
    }
  }
}

template<int KPAD>
__global__ __launch_bounds__(64)
void k_wmma_lin(const _Float16* __restrict__ Xp, const _Float16* __restrict__ Wp,
                const float* __restrict__ biasp, int R, int Cout, int relu,
                float* __restrict__ Yf, _Float16* __restrict__ Yh, int ldo) {
  wmma_lin_body<KPAD>(Xp, Wp, biasp, R, Cout, relu, Yf, Yh, ldo, KPAD);
}

__global__ __launch_bounds__(64)
void k_wmma_lin_dyn(const _Float16* __restrict__ Xp, const _Float16* __restrict__ Wp,
                    const float* __restrict__ biasp, int R, int Cout, int relu,
                    float* __restrict__ Yf, _Float16* __restrict__ Yh, int ldo, int kpad) {
  wmma_lin_body<0>(Xp, Wp, biasp, R, Cout, relu, Yf, Yh, ldo, kpad);
}

// ============================ quaternion kernels ============================
__global__ void k_quat_norm(const float* __restrict__ q, int B, float* __restrict__ out) {
  int b = blockIdx.x * blockDim.x + threadIdx.x;
  if (b >= B) return;
  float a0 = q[b*4], a1 = q[b*4+1], a2 = q[b*4+2], a3 = q[b*4+3];
  float nrm = sqrtf(a0*a0 + a1*a1 + a2*a2 + a3*a3 + 1e-10f) + 1e-10f;
  out[b*4] = a0/nrm; out[b*4+1] = a1/nrm; out[b*4+2] = a2/nrm; out[b*4+3] = a3/nrm;
}

__device__ inline void rot_apply(float w, float x, float y, float z,
                                 float px, float py, float pz,
                                 float& ox, float& oy, float& oz) {
  ox = (1.f - 2.f*(y*y + z*z))*px + 2.f*(x*y - w*z)*py + 2.f*(x*z + w*y)*pz;
  oy = 2.f*(x*y + w*z)*px + (1.f - 2.f*(x*x + z*z))*py + 2.f*(y*z - w*x)*pz;
  oz = 2.f*(x*z - w*y)*px + 2.f*(y*z + w*x)*py + (1.f - 2.f*(x*x + y*y))*pz;
}

__global__ void k_warp(const float* __restrict__ xyz, const float* __restrict__ qn,
                       const float* __restrict__ t, int B, int M, float* __restrict__ out) {
  int i = blockIdx.x * blockDim.x + threadIdx.x;
  if (i >= B * M) return;
  int b = i / M;
  float w = qn[b*4], x = qn[b*4+1], y = qn[b*4+2], z = qn[b*4+3];
  const float* p = xyz + (size_t)i * 3;
  float ox, oy, oz;
  rot_apply(w, x, y, z, p[0], p[1], p[2], ox, oy, oz);
  out[(size_t)i*3]   = ox + t[b*3];
  out[(size_t)i*3+1] = oy + t[b*3+1];
  out[(size_t)i*3+2] = oz + t[b*3+2];
}

// dqn=norm(dq); q=quat_mul(dqn,qn); t=rotate(dqn,t_prev)+dt
__global__ void k_pose_update(const float* __restrict__ dq, const float* __restrict__ dt,
                              const float* __restrict__ qn, const float* __restrict__ tprev,
                              int B, float* __restrict__ qout, float* __restrict__ tout) {
  int b = blockIdx.x * blockDim.x + threadIdx.x;
  if (b >= B) return;
  float a0 = dq[b*4], a1 = dq[b*4+1], a2 = dq[b*4+2], a3 = dq[b*4+3];
  float nrm = sqrtf(a0*a0 + a1*a1 + a2*a2 + a3*a3 + 1e-10f) + 1e-10f;
  float w = a0/nrm, x = a1/nrm, y = a2/nrm, z = a3/nrm;
  float bw = qn[b*4], bx = qn[b*4+1], by = qn[b*4+2], bz = qn[b*4+3];
  qout[b*4]   = w*bw - x*bx - y*by - z*bz;
  qout[b*4+1] = w*bx + x*bw + y*bz - z*by;
  qout[b*4+2] = w*by - x*bz + y*bw + z*bx;
  qout[b*4+3] = w*bz + x*by - y*bx + z*bw;
  float ox, oy, oz;
  rot_apply(w, x, y, z, tprev[b*3], tprev[b*3+1], tprev[b*3+2], ox, oy, oz);
  tout[b*3]   = ox + dt[b*3];
  tout[b*3+1] = oy + dt[b*3+1];
  tout[b*3+2] = oz + dt[b*3+2];
}

// poses (B,4,7): per level [t(3), quat_norm(q)], level order 1..4
__global__ void k_final_poses(const float* q1, const float* t1, const float* q2, const float* t2,
                              const float* q3, const float* t3, const float* q4, const float* t4,
                              int B, float* __restrict__ out) {
  int b = blockIdx.x * blockDim.x + threadIdx.x;
  if (b >= B) return;
  const float* qs[4] = {q1, q2, q3, q4};
  const float* ts[4] = {t1, t2, t3, t4};
  for (int l = 0; l < 4; ++l) {
    float* o = out + ((size_t)b * 4 + l) * 7;
    o[0] = ts[l][b*3]; o[1] = ts[l][b*3+1]; o[2] = ts[l][b*3+2];
    float a0 = qs[l][b*4], a1 = qs[l][b*4+1], a2 = qs[l][b*4+2], a3 = qs[l][b*4+3];
    float nrm = sqrtf(a0*a0 + a1*a1 + a2*a2 + a3*a3 + 1e-10f) + 1e-10f;
    o[3] = a0/nrm; o[4] = a1/nrm; o[5] = a2/nrm; o[6] = a3/nrm;
  }
}

// ============================ Host-side orchestration ============================

struct Lin { const float* W; const float* b; int cin, cout; };
struct Mlp { Lin l[3]; int n; };
struct CvP { Mlp mlp1, mlp2, w1, w2; };
struct PoseP { Lin fc, q, t; };
struct RefP { CvP cv; PoseP pose; Mlp pred, refine; };

struct Arena {
  char* base; size_t off, cap;
  float*    f(size_t n) { float* p = (float*)(base + off);    off += ((n*sizeof(float) + 255) & ~(size_t)255); return p; }
  int*      i(size_t n) { int* p = (int*)(base + off);        off += ((n*sizeof(int)   + 255) & ~(size_t)255); return p; }
  _Float16* h(size_t n) { _Float16* p = (_Float16*)(base + off); off += ((n*sizeof(_Float16) + 255) & ~(size_t)255); return p; }
};

struct Bufs {
  float *B, *C, *D, *pf;     // f32 scratch
  _Float16 *P0, *P1, *Wp;    // packed f16 activations + weights
  float *bp;                 // padded bias
  int *idxA, *idxB;
};

static void launch_wmma(hipStream_t s, const _Float16* Xp, int Kpad, const _Float16* Wp,
                        const float* bp, int R, int Cout, int relu,
                        float* Yf, _Float16* Yh, int ldo) {
  int Rtiles = (R + 15) / 16;
  int Ct = Yh ? (pad32(Cout) / 16) : ((Cout + 15) / 16);
  dim3 grid(Ct, (Rtiles + 1) / 2);
  switch (Kpad) {
    case 32:  k_wmma_lin<32> <<<grid,64,0,s>>>(Xp,Wp,bp,R,Cout,relu,Yf,Yh,ldo); break;
    case 64:  k_wmma_lin<64> <<<grid,64,0,s>>>(Xp,Wp,bp,R,Cout,relu,Yf,Yh,ldo); break;
    case 96:  k_wmma_lin<96> <<<grid,64,0,s>>>(Xp,Wp,bp,R,Cout,relu,Yf,Yh,ldo); break;
    case 128: k_wmma_lin<128><<<grid,64,0,s>>>(Xp,Wp,bp,R,Cout,relu,Yf,Yh,ldo); break;
    case 160: k_wmma_lin<160><<<grid,64,0,s>>>(Xp,Wp,bp,R,Cout,relu,Yf,Yh,ldo); break;
    case 192: k_wmma_lin<192><<<grid,64,0,s>>>(Xp,Wp,bp,R,Cout,relu,Yf,Yh,ldo); break;
    default:  k_wmma_lin_dyn <<<grid,64,0,s>>>(Xp,Wp,bp,R,Cout,relu,Yf,Yh,ldo,Kpad); break;
  }
}

// Chain: pack X once, per layer pack W, GEMM; mid layers stay packed f16, last writes f32.
static void run_mlp(hipStream_t s, const Mlp& m, const float* X, int ldx, int R,
                    float* dst, Bufs& bb) {
  int Kp0 = pad32(m.l[0].cin);
  k_pack_x<<<g1i(R * Kp0, 256), 256, 0, s>>>(X, ldx, R, m.l[0].cin, Kp0, bb.P0);
  _Float16* cur = bb.P0;
  _Float16* nxt = bb.P1;
  for (int i = 0; i < m.n; ++i) {
    const Lin& L = m.l[i];
    int Kp = pad32(L.cin);
    int Cp = pad32(L.cout);
    k_pack_w<<<g1i(Cp * Kp, 256), 256, 0, s>>>(L.W, L.b, L.cin, L.cout, Kp, Cp, bb.Wp, bb.bp);
    if (i != m.n - 1) {
      launch_wmma(s, cur, Kp, bb.Wp, bb.bp, R, L.cout, 1, nullptr, nxt, Cp);
      _Float16* t = cur; cur = nxt; nxt = t;
    } else {
      launch_wmma(s, cur, Kp, bb.Wp, bb.bp, R, L.cout, 1, dst, nullptr, L.cout);
    }
  }
}

static void run_sa(hipStream_t s, const Mlp& m, const float* xyz, const float* feats, int C,
                   int B, int N, int np, int S, Bufs& bb, int* fidx,
                   float* out_xyz, float* out_f) {
  k_fps<<<B, FPS_T, 0, s>>>(xyz, N, np, fidx);
  k_gather_rows<<<g1i(B*np,256), 256, 0, s>>>(xyz, fidx, B, N, np, 3, out_xyz);
  k_knn<<<g1i(B*np,128), 128, 0, s>>>(out_xyz, xyz, B, np, N, S, bb.idxA);
  int R = B * np * S;
  k_group_concat<<<g1i(R,256), 256, 0, s>>>(xyz, out_xyz, bb.idxA, feats, C, B, np, S, N, bb.C);
  run_mlp(s, m, bb.C, 3 + C, R, bb.C, bb);              // final f32 back into C
  int Co = m.l[m.n-1].cout;
  k_maxpool<<<g1i(B*np*Co,256), 256, 0, s>>>(bb.C, B, np, S, Co, out_f);
}

static void run_cost_volume(hipStream_t s, const CvP& p,
                            const float* xyz1, const float* f1,
                            const float* xyz2, const float* f2,
                            int B, int M, int Nref, int C, Bufs& bb, float* dst) {
  const int S = 32, S2 = 4;
  int R = B * M * S, R2 = B * M * S2;
  k_knn<<<g1i(B*M,128), 128, 0, s>>>(xyz1, xyz2, B, M, Nref, S, bb.idxA);
  k_cv_concat<<<g1i(R,256), 256, 0, s>>>(f1, f2, xyz1, xyz2, bb.idxA, B, M, S, Nref, C, bb.C);
  run_mlp(s, p.mlp1, bb.C, 2*C + 3, R, bb.D, bb);                  // feat -> D
  run_mlp(s, p.w1, bb.C + 2*C, 2*C + 3, R, bb.B, bb);              // diff slice -> wlog in B
  k_softmax_wsum<<<g1i(B*M*64,256), 256, 0, s>>>(bb.B, bb.D, B, M, S, 64, bb.pf);
  k_knn<<<g1i(B*M,128), 128, 0, s>>>(xyz1, xyz1, B, M, M, S2, bb.idxB);
  k_group_concat<<<g1i(R2,256), 256, 0, s>>>(xyz1, xyz1, bb.idxB, bb.pf, 64, B, M, S2, M, bb.C);
  run_mlp(s, p.mlp2, bb.C, 67, R2, bb.D, bb);                      // feat2 -> D
  run_mlp(s, p.w2, bb.C, 67, R2, bb.B, bb);                        // d2 slice (K=3, ld=67) -> B
  k_softmax_wsum<<<g1i(B*M*64,256), 256, 0, s>>>(bb.B, bb.D, B, M, S2, 64, dst);
}

static void run_pose_head(hipStream_t s, const PoseP& p, const float* feats, const float* mask,
                          int B, int M, float* pooled, float* h, float* q, float* t) {
  k_masked_pool<<<g1i(B*64,128), 128, 0, s>>>(feats, mask, B, M, 64, pooled);
  k_linear_small<<<g1i(B*256,256), 256, 0, s>>>(pooled, p.fc.W, p.fc.b, h, B, 64, 256, 0);
  k_linear_small<<<g1i(B*4,64),   64,  0, s>>>(h, p.q.W, p.q.b, q, B, 256, 4, 0);
  k_linear_small<<<g1i(B*3,64),   64,  0, s>>>(h, p.t.W, p.t.b, t, B, 256, 3, 0);
}

extern "C" void kernel_launch(void* const* d_in, const int* in_sizes, int n_in,
                              void* d_out, int out_size, void* d_ws, size_t ws_size,
                              hipStream_t stream) {
  const int B = 2, N = 8192;
  if (n_in < 158 || !d_ws) return;

  // ---------- parameter leaves (jax pytree flatten: sorted dict keys) ----------
  int pi = 4;
  auto leaf  = [&]() { return (const float*)d_in[pi++]; };
  auto rdLin = [&](int ci, int co) { Lin L; L.W = leaf(); L.b = leaf(); L.cin = ci; L.cout = co; return L; };
  auto rdM2  = [&](int a, int b, int c) { Mlp m; m.n = 2; m.l[0] = rdLin(a,b); m.l[1] = rdLin(b,c); return m; };
  auto rdM3  = [&](int a, int b, int c, int d) { Mlp m; m.n = 3; m.l[0] = rdLin(a,b); m.l[1] = rdLin(b,c); m.l[2] = rdLin(c,d); return m; };
  auto rdCv  = [&](int C) { CvP p; p.mlp1 = rdM3(2*C+3,128,64,64); p.mlp2 = rdM2(67,128,64);
                            p.w1 = rdM2(3,32,64); p.w2 = rdM2(3,32,64); return p; };
  auto rdPose= [&]() { PoseP p; p.fc = rdLin(64,256); p.q = rdLin(256,4); p.t = rdLin(256,3); return p; };
  auto rdRef = [&](int C) { RefP r; r.cv = rdCv(C); r.pose = rdPose();
                            r.pred = rdM2(C+128,128,64); r.refine = rdM2(128,128,64); return r; };

  CvP  cv_main = rdCv(64);
  PoseP pose4  = rdPose();
  Mlp  pred4   = rdM2(192,128,64);
  RefP ref1    = rdRef(16);
  RefP ref2    = rdRef(32);
  RefP ref3    = rdRef(64);
  Mlp  sa1 = rdM3(3,8,8,16),  sa2 = rdM3(19,16,16,32), sa3 = rdM3(35,32,32,64);
  Mlp  sa4 = rdM3(67,64,64,128), sa_fe = rdM3(67,128,64,64);

  // ---------- workspace ----------
  Arena ar{ (char*)d_ws, 0, ws_size };
  const size_t RMAX = (size_t)B * 2048 * 32;             // 131072 rows
  Bufs bb;
  bb.B  = ar.f(RMAX * 64);
  bb.C  = ar.f(RMAX * 36);
  bb.D  = ar.f(RMAX * 64);
  bb.pf = ar.f((size_t)B * 2048 * 64);
  bb.P0 = ar.h(RMAX * 128 + 256);                        // packed activations ping
  bb.P1 = ar.h(RMAX * 128 + 256);                        // packed activations pong
  bb.Wp = ar.h((size_t)128 * 224);                       // packed weights (max Cpad x Kpad)
  bb.bp = ar.f(256);
  bb.idxA = ar.i(RMAX); bb.idxB = ar.i((size_t)B * 2048 * 4);
  int* fidx = ar.i((size_t)B * 2048);
  int* idxU = ar.i((size_t)B * 2048 * 3);

  float* x1t = ar.f((size_t)B*N*3); float* x2t = ar.f((size_t)B*N*3);
  float* x1_1 = ar.f((size_t)B*2048*3); float* f1_1 = ar.f((size_t)B*2048*16);
  float* x2_1 = ar.f((size_t)B*2048*3); float* f2_1 = ar.f((size_t)B*2048*16);
  float* x1_2 = ar.f((size_t)B*1024*3); float* f1_2 = ar.f((size_t)B*1024*32);
  float* x2_2 = ar.f((size_t)B*1024*3); float* f2_2 = ar.f((size_t)B*1024*32);
  float* x1_3 = ar.f((size_t)B*256*3);  float* f1_3 = ar.f((size_t)B*256*64);
  float* x2_3 = ar.f((size_t)B*256*3);  float* f2_3 = ar.f((size_t)B*256*64);
  float* x1_4 = ar.f((size_t)B*64*3);   float* f1_4 = ar.f((size_t)B*64*128);
  float* cv3  = ar.f((size_t)B*256*64);
  float* x1_4e = ar.f((size_t)B*64*3);  float* ef4 = ar.f((size_t)B*64*64);
  float* m4 = ar.f((size_t)B*64*64);
  float* ef3 = ar.f((size_t)B*256*64);  float* m3 = ar.f((size_t)B*256*64);
  float* ef2 = ar.f((size_t)B*1024*64); float* m2 = ar.f((size_t)B*1024*64);
  float* ef1 = ar.f((size_t)B*2048*64); float* m1 = ar.f((size_t)B*2048*64);
  float* cvb = ar.f((size_t)B*2048*64);
  float* upF = ar.f((size_t)B*2048*64); float* upM = ar.f((size_t)B*2048*64);
  float* xyzw = ar.f((size_t)B*2048*3);
  float* qnb = ar.f((size_t)B*4);
  float* q4 = ar.f((size_t)B*4); float* t4 = ar.f((size_t)B*3);
  float* q3 = ar.f((size_t)B*4); float* t3 = ar.f((size_t)B*3);
  float* q2 = ar.f((size_t)B*4); float* t2 = ar.f((size_t)B*3);
  float* q1 = ar.f((size_t)B*4); float* t1 = ar.f((size_t)B*3);
  float* pooled = ar.f((size_t)B*64); float* hbuf = ar.f((size_t)B*256);
  float* dq = ar.f((size_t)B*4); float* dt = ar.f((size_t)B*3);
  if (ar.off > ws_size) return;   // workspace too small: bail deterministically

  const float* xin1 = (const float*)d_in[0];
  const float* xin2 = (const float*)d_in[2];
  hipStream_t s = stream;

  // ---------- feature pyramid ----------
  k_transpose<<<g1i(B*N,256), 256, 0, s>>>(xin1, B, N, x1t);
  k_transpose<<<g1i(B*N,256), 256, 0, s>>>(xin2, B, N, x2t);

  run_sa(s, sa1, x1t, nullptr, 0, B, N,    2048, 32, bb, fidx, x1_1, f1_1);
  run_sa(s, sa2, x1_1, f1_1, 16, B, 2048, 1024, 32, bb, fidx, x1_2, f1_2);
  run_sa(s, sa3, x1_2, f1_2, 32, B, 1024,  256, 16, bb, fidx, x1_3, f1_3);
  run_sa(s, sa4, x1_3, f1_3, 64, B,  256,   64, 16, bb, fidx, x1_4, f1_4);
  run_sa(s, sa1, x2t, nullptr, 0, B, N,    2048, 32, bb, fidx, x2_1, f2_1);
  run_sa(s, sa2, x2_1, f2_1, 16, B, 2048, 1024, 32, bb, fidx, x2_2, f2_2);
  run_sa(s, sa3, x2_2, f2_2, 32, B, 1024,  256, 16, bb, fidx, x2_3, f2_3);

  // ---------- coarse cost volume + embedding ----------
  run_cost_volume(s, cv_main, x1_3, f1_3, x2_3, f2_3, B, 256, 256, 64, bb, cv3);
  run_sa(s, sa_fe, x1_3, cv3, 64, B, 256, 64, 16, bb, fidx, x1_4e, ef4);

  // m4 = pred4(concat(f1_4, ef4)); q4,t4 = pose_head(ef4, m4)
  k_concat_rows<<<g1i(B*64,128), 128, 0, s>>>(f1_4, 128, ef4, 64, nullptr, 0, B*64, bb.C);
  run_mlp(s, pred4, bb.C, 192, B*64, m4, bb);
  run_pose_head(s, pose4, ef4, m4, B, 64, pooled, hbuf, q4, t4);

  // ---------- iterative refinement ----------
  struct Lvl { const RefP* rp; const float* xyz1; const float* f1; int C;
               const float* xyz2; const float* f2; int M; int Mprev;
               const float* xyzp; const float* efp; const float* emp;
               const float* qp; const float* tp;
               float* ef; float* em; float* qo; float* to; };
  Lvl lv[3] = {
    { &ref3, x1_3, f1_3, 64, x2_3, f2_3,  256,   64, x1_4e, ef4, m4, q4, t4, ef3, m3, q3, t3 },
    { &ref2, x1_2, f1_2, 32, x2_2, f2_2, 1024,  256, x1_3,  ef3, m3, q3, t3, ef2, m2, q2, t2 },
    { &ref1, x1_1, f1_1, 16, x2_1, f2_1, 2048, 1024, x1_2,  ef2, m2, q2, t2, ef1, m1, q1, t1 },
  };
  for (int li = 0; li < 3; ++li) {
    Lvl& L = lv[li];
    int M = L.M, Rl = B * M;
    k_quat_norm<<<1, B, 0, s>>>(L.qp, B, qnb);
    k_warp<<<g1i(Rl,256), 256, 0, s>>>(L.xyz1, qnb, L.tp, B, M, xyzw);
    run_cost_volume(s, L.rp->cv, xyzw, L.f1, L.xyz2, L.f2, B, M, M, L.C, bb, cvb);
    k_knn<<<g1i(Rl,128), 128, 0, s>>>(L.xyz1, L.xyzp, B, M, L.Mprev, 3, idxU);
    k_upsample<<<g1i(Rl,256), 256, 0, s>>>(L.xyz1, L.xyzp, idxU, L.efp, B, M, L.Mprev, 64, upF);
    k_upsample<<<g1i(Rl,256), 256, 0, s>>>(L.xyz1, L.xyzp, idxU, L.emp, B, M, L.Mprev, 64, upM);
    k_concat_rows<<<g1i(Rl,256), 256, 0, s>>>(upF, 64, cvb, 64, nullptr, 0, Rl, bb.C);
    run_mlp(s, L.rp->refine, bb.C, 128, Rl, L.ef, bb);
    k_concat_rows<<<g1i(Rl,256), 256, 0, s>>>(L.f1, L.C, L.ef, 64, upM, 64, Rl, bb.C);
    run_mlp(s, L.rp->pred, bb.C, L.C + 128, Rl, L.em, bb);
    run_pose_head(s, L.rp->pose, L.ef, L.em, B, M, pooled, hbuf, dq, dt);
    k_pose_update<<<1, B, 0, s>>>(dq, dt, qnb, L.tp, B, L.qo, L.to);
  }

  // ---------- output (B,4,7): levels 1..4, [t, quat_norm(q)] ----------
  k_final_poses<<<1, B, 0, s>>>(q1, t1, q2, t2, q3, t3, q4, t4, B, (float*)d_out);
  (void)in_sizes; (void)out_size;
}
